// multihead_attention_35802847380078
// MI455X (gfx1250) — matrix operand order repaired and checked by reading, not compiled, not run on hardware
//
#include <hip/hip_runtime.h>
#include <hip/hip_bf16.h>

// ---------------------------------------------------------------------------
// Fused transformer block for MI455X (gfx1250, wave32, WMMA + TDM).
// B=8, T=1024, U=1024, H=8, d=128.
// ---------------------------------------------------------------------------

typedef __attribute__((ext_vector_type(16))) _Float16 v16h;
typedef __attribute__((ext_vector_type(8)))  _Float16 v8h;
typedef __attribute__((ext_vector_type(8)))  float    v8f;
typedef __attribute__((ext_vector_type(4)))  float    v4f;
typedef __attribute__((ext_vector_type(4)))  unsigned int u32x4;
typedef __attribute__((ext_vector_type(8)))  unsigned int u32x8;

#define WMMA_F16(A, B, C) \
  __builtin_amdgcn_wmma_f32_16x16x32_f16(false, (A), false, (B), (short)0, (C), false, false)

#define NEG_MASK (-4294967295.0f)
#define INV_SQRT_D 0.08838834764831845f
#define U_DIM 1024
#define M_DIM 8192  // B*T

// ---------------------------------------------------------------------------
// TDM: DMA a 2D tile (128 rows x 64 bytes) from global into LDS (packed
// row-major, 64B pitch). Descriptor per CDNA5 ISA.
//   tensor: rows of Wt (1024 x 1024 f16) viewed as 4-byte elements:
//     tensor_dim0 = 512 dwords/row, dim0_stride = 512, tensor_dim1 = 1024 rows
//   tile: tile_dim0 = 16 dwords (32 halves), tile_dim1 = 128 rows
// ---------------------------------------------------------------------------
__device__ __forceinline__ void tdm_load_b_tile(const _Float16* gsrc, unsigned lds_off) {
  unsigned long long ga = (unsigned long long)(size_t)gsrc;
  u32x4 g0;
  g0[0] = __builtin_amdgcn_readfirstlane(1u);                      // count=1, user mode
  g0[1] = __builtin_amdgcn_readfirstlane(lds_off);                 // lds_addr
  g0[2] = __builtin_amdgcn_readfirstlane((unsigned)ga);            // global_addr[31:0]
  g0[3] = __builtin_amdgcn_readfirstlane(
            (unsigned)(ga >> 32) | 0x80000000u);                   // addr[56:32] | type=2
  u32x8 g1;
  g1[0] = 0x00020000u;                       // workgroup_mask=0, data_size=2 (4B)
  g1[1] = (512u & 0xffffu) << 16;            // tensor_dim0[15:0] -> bits 63:48
  g1[2] = (512u >> 16) | ((1024u & 0xffffu) << 16);  // dim0 hi | tensor_dim1 lo
  g1[3] = (1024u >> 16) | (16u << 16);       // dim1 hi | tile_dim0=16 dwords
  g1[4] = 128u;                              // tile_dim1=128, tile_dim2=0
  g1[5] = 512u;                              // tensor_dim0_stride[31:0]
  g1[6] = 0u;                                // stride0 hi | dim1_stride lo (unused, 2D)
  g1[7] = 0u;
  asm volatile("tensor_load_to_lds %0, %1" :: "s"(g0), "s"(g1) : "memory");
}

// ---------------------------------------------------------------------------
// Kernel 1: W (f32, row-major KxN) -> Wt (f16, row-major NxK)
// ---------------------------------------------------------------------------
__global__ __launch_bounds__(256) void wtrans_kernel(const float* __restrict__ W,
                                                     _Float16* __restrict__ Wt) {
  int idx = blockIdx.x * 256 + threadIdx.x;   // 1M threads
  int n = idx >> 10;
  int k = idx & 1023;
  Wt[idx] = (_Float16)W[(size_t)k * U_DIM + n];
}

// ---------------------------------------------------------------------------
// Kernel 2: key/query padding masks: sign(|rowsum|)
// ---------------------------------------------------------------------------
__global__ __launch_bounds__(256) void mask_kernel(const float* __restrict__ queries,
                                                   const float* __restrict__ keys,
                                                   float* __restrict__ qmask,
                                                   float* __restrict__ kmask) {
  int row = blockIdx.x;
  int t = threadIdx.x;
  int lane = t & 31, wid = t >> 5;
  const float* src = (row < M_DIM) ? (queries + (size_t)row * U_DIM)
                                   : (keys + (size_t)(row - M_DIM) * U_DIM);
  float s = 0.f;
#pragma unroll
  for (int i = 0; i < 4; ++i) s += src[t + i * 256];
#pragma unroll
  for (int off = 1; off < 32; off <<= 1) s += __shfl_xor(s, off);
  __shared__ float sh[8];
  if (lane == 0) sh[wid] = s;
  __syncthreads();
  if (t == 0) {
    float tot = 0.f;
    for (int w = 0; w < 8; ++w) tot += sh[w];
    float* dst = (row < M_DIM) ? qmask : kmask;
    dst[row & (M_DIM - 1)] = (tot != 0.f) ? 1.f : 0.f;
  }
}

// ---------------------------------------------------------------------------
// Kernel 3: projection GEMM  out = relu(X @ W + b), f16 output.
//   A (f32 activations): direct global b128 loads + cvt into A-fragments.
//   B (f16 Wt): TDM tensor_load_to_lds, double-buffered, ds_load fragments.
// Block: 256 thr = 8 waves (4x2), block tile 128x128, wave tile 32x64, K step 32.
// ---------------------------------------------------------------------------
__global__ __launch_bounds__(256) void proj_gemm_kernel(const float* __restrict__ X,
                                                        const _Float16* __restrict__ Wt,
                                                        const float* __restrict__ bias,
                                                        _Float16* __restrict__ out,
                                                        long long out_rs, long long out_cs) {
  __shared__ __align__(64) _Float16 btile[2][128 * 32];  // 2 x 8KB B tiles
  const int lane = threadIdx.x & 31;
  const int wid  = threadIdx.x >> 5;
  const int wm = wid >> 1, wn = wid & 1;
  const int l16 = lane & 15, hi = lane >> 4;
  const int m_base = blockIdx.y * 128 + wm * 32;
  const int n_blk  = blockIdx.x * 128;
  const int n_base = n_blk + wn * 64;

  v8f acc[2][4];
#pragma unroll
  for (int i = 0; i < 2; ++i)
#pragma unroll
    for (int j = 0; j < 4; ++j) acc[i][j] = {};

  const unsigned lds0 = (unsigned)(size_t)(void*)&btile[0][0];
  const unsigned lds1 = (unsigned)(size_t)(void*)&btile[1][0];

  // Prologue: DMA tile for k0 = 0.
  if (wid == 0) {
    tdm_load_b_tile(Wt + (size_t)n_blk * U_DIM, lds0);
  }

  const int NITER = U_DIM / 32;
  for (int it = 0; it < NITER; ++it) {
    const int k0 = it * 32;
    if (wid == 0) {
      if (it + 1 < NITER) {
        // Issue next tile (safe: end-of-iter barrier of it-1 retired all
        // readers of that buffer), then wait for the current tile only.
        tdm_load_b_tile(Wt + (size_t)n_blk * U_DIM + (k0 + 32),
                        (it & 1) ? lds0 : lds1);
        __builtin_amdgcn_s_wait_tensorcnt(1);
      } else {
        __builtin_amdgcn_s_wait_tensorcnt(0);
      }
    }
    __syncthreads();  // current B tile visible to all waves

    // A fragments: f32 -> f16 on the fly. row = l16, halves
    // {kb..kb+15}, kb = 16*hi.
    v16h fa[2];
#pragma unroll
    for (int i = 0; i < 2; ++i) {
      const float* ap = X + (size_t)(m_base + i * 16 + l16) * U_DIM + k0 + hi * 16;
      v4f x0 = *(const v4f*)(ap + 0);
      v4f x1 = *(const v4f*)(ap + 4);
      v4f x2 = *(const v4f*)(ap + 8);
      v4f x3 = *(const v4f*)(ap + 12);
#pragma unroll
      for (int j = 0; j < 4; ++j) {
        fa[i][j]      = (_Float16)x0[j];
        fa[i][4 + j]  = (_Float16)x1[j];
        fa[i][8 + j]  = (_Float16)x2[j];
        fa[i][12 + j] = (_Float16)x3[j];
      }
    }
    // B fragments from LDS: col n = l16 (tile row), 32B contiguous K run.
    const _Float16* bt = &btile[it & 1][0];
#pragma unroll
    for (int j = 0; j < 4; ++j) {
      const _Float16* bp = bt + (wn * 64 + j * 16 + l16) * 32 + hi * 16;
      v16h fb = *(const v16h*)bp;
#pragma unroll
      for (int i = 0; i < 2; ++i) acc[i][j] = WMMA_F16(fa[i], fb, acc[i][j]);
    }
    __syncthreads();  // all waves done reading this buffer before it is reused
  }

  // Epilogue: bias + ReLU + f16 store. C layout: row = v + 8*hi, col = l16.
#pragma unroll
  for (int j = 0; j < 4; ++j) {
    int n = n_base + j * 16 + l16;
    float bv = bias[n];
#pragma unroll
    for (int i = 0; i < 2; ++i) {
#pragma unroll
      for (int v = 0; v < 8; ++v) {
        int m = m_base + i * 16 + v + 8 * hi;
        float val = acc[i][j][v] + bv;
        val = fmaxf(val, 0.f);
        out[(size_t)m * out_rs + (size_t)n * out_cs] = (_Float16)val;
      }
    }
  }
}

// ---------------------------------------------------------------------------
// Kernel 4: flash attention, one wave per (head, batch, 16-query tile).
//   Qp,Kp: f16 [8192 x 1024] (head h occupies cols [h*128,(h+1)*128))
//   Vt:    f16 [1024 x 8192], Vt[u][m] = V[m][u]
// ---------------------------------------------------------------------------
__global__ __launch_bounds__(256) void flash_attn_kernel(const _Float16* __restrict__ Qp,
                                                         const _Float16* __restrict__ Kp,
                                                         const _Float16* __restrict__ Vt,
                                                         const float* __restrict__ kmask,
                                                         const float* __restrict__ qmask,
                                                         float* __restrict__ attn_out) {
  __shared__ _Float16 plds[8][16 * 64];  // per-wave 16x64 P tile (2KB each)
  const int lane = threadIdx.x & 31;
  const int wid  = threadIdx.x >> 5;
  const int l16 = lane & 15, hi = lane >> 4;

  const int gw = blockIdx.x * 8 + wid;  // 4096 waves total
  const int qt = gw & 63;
  const int b  = (gw >> 6) & 7;
  const int h  = gw >> 9;
  const int q0 = qt * 16;

  // Hoist Q A-fragments for the whole 16x128 Q tile (4 d-steps).
  v16h fq[4];
  {
    const _Float16* qbase = Qp + (size_t)(b * 1024 + q0 + l16) * U_DIM + h * 128;
#pragma unroll
    for (int ds = 0; ds < 4; ++ds) {
      const _Float16* p = qbase + ds * 32 + hi * 16;
      v8h a0 = *(const v8h*)p;
      v8h a1 = *(const v8h*)(p + 8);
#pragma unroll
      for (int e = 0; e < 8; ++e) { fq[ds][e] = a0[e]; fq[ds][8 + e] = a1[e]; }
    }
  }

  float m8[8], l8[8];
  v8f o[8];
#pragma unroll
  for (int v = 0; v < 8; ++v) { m8[v] = -3.0e38f; l8[v] = 0.f; }
#pragma unroll
  for (int nt = 0; nt < 8; ++nt) o[nt] = {};

  for (int kb0 = 0; kb0 < q0 + 16; kb0 += 64) {
    // ---- scores: S = Q Kt / sqrt(d) over a 16x64 K chunk ----
    v8f s[4];
#pragma unroll
    for (int j = 0; j < 4; ++j) s[j] = {};
#pragma unroll
    for (int j = 0; j < 4; ++j) {
#pragma unroll
      for (int ds = 0; ds < 4; ++ds) {
        const _Float16* kp = Kp + (size_t)(b * 1024 + kb0 + j * 16 + l16) * U_DIM
                             + h * 128 + ds * 32 + hi * 16;
        v16h fk = *(const v16h*)kp;
        s[j] = WMMA_F16(fq[ds], fk, s[j]);
      }
    }

    // ---- scale + causal mask + key mask; per-row chunk max ----
    float tmax[8];
#pragma unroll
    for (int v = 0; v < 8; ++v) tmax[v] = -3.0e38f;
#pragma unroll
    for (int j = 0; j < 4; ++j) {
      int kc = kb0 + j * 16 + l16;
      float km = kmask[b * 1024 + kc];
#pragma unroll
      for (int v = 0; v < 8; ++v) {
        int qr = q0 + v + 8 * hi;
        float val = s[j][v] * INV_SQRT_D;
        if (kc > qr || km == 0.f) val = NEG_MASK;
        s[j][v] = val;
        tmax[v] = fmaxf(tmax[v], val);
      }
    }
#pragma unroll
    for (int v = 0; v < 8; ++v) {
      tmax[v] = fmaxf(tmax[v], __shfl_xor(tmax[v], 1));
      tmax[v] = fmaxf(tmax[v], __shfl_xor(tmax[v], 2));
      tmax[v] = fmaxf(tmax[v], __shfl_xor(tmax[v], 4));
      tmax[v] = fmaxf(tmax[v], __shfl_xor(tmax[v], 8));
    }

    // ---- online softmax update ----
    float sc8[8];
#pragma unroll
    for (int v = 0; v < 8; ++v) {
      float mn = fmaxf(m8[v], tmax[v]);
      sc8[v] = __expf(m8[v] - mn);
      m8[v] = mn;
      l8[v] *= sc8[v];
    }
#pragma unroll
    for (int nt = 0; nt < 8; ++nt)
#pragma unroll
      for (int v = 0; v < 8; ++v) o[nt][v] *= sc8[v];

    float rs[8];
#pragma unroll
    for (int v = 0; v < 8; ++v) rs[v] = 0.f;
#pragma unroll
    for (int j = 0; j < 4; ++j)
#pragma unroll
      for (int v = 0; v < 8; ++v) {
        float p = __expf(s[j][v] - m8[v]);
        s[j][v] = p;
        rs[v] += p;
      }
#pragma unroll
    for (int v = 0; v < 8; ++v) {
      rs[v] += __shfl_xor(rs[v], 1);
      rs[v] += __shfl_xor(rs[v], 2);
      rs[v] += __shfl_xor(rs[v], 4);
      rs[v] += __shfl_xor(rs[v], 8);
      l8[v] += rs[v];
    }

    // ---- P (C layout) -> LDS -> A layout fragments ----
    _Float16* pl = plds[wid];
#pragma unroll
    for (int j = 0; j < 4; ++j)
#pragma unroll
      for (int v = 0; v < 8; ++v)
        pl[(v + 8 * hi) * 64 + j * 16 + l16] = (_Float16)s[j][v];

    asm volatile("s_wait_dscnt 0" ::: "memory");

    v16h fp[2];
#pragma unroll
    for (int ks = 0; ks < 2; ++ks) {
      const _Float16* p = pl + l16 * 64 + ks * 32 + hi * 16;
      v8h a0 = *(const v8h*)p;
      v8h a1 = *(const v8h*)(p + 8);
#pragma unroll
      for (int e = 0; e < 8; ++e) { fp[ks][e] = a0[e]; fp[ks][8 + e] = a1[e]; }
    }

    // ---- O += P @ V (B fragments straight from transposed V) ----
#pragma unroll
    for (int nt = 0; nt < 8; ++nt) {
#pragma unroll
      for (int ks = 0; ks < 2; ++ks) {
        const _Float16* vp = Vt + (size_t)(h * 128 + nt * 16 + l16) * M_DIM
                             + b * 1024 + kb0 + ks * 32 + hi * 16;
        v16h fv = *(const v16h*)vp;
        o[nt] = WMMA_F16(fp[ks], fv, o[nt]);
      }
    }
  }

  // ---- finalize: /l, * query_mask, store merged-head f32 ----
  float inv[8];
#pragma unroll
  for (int v = 0; v < 8; ++v) {
    int qr = q0 + v + 8 * hi;
    inv[v] = qmask[b * 1024 + qr] / l8[v];
  }
#pragma unroll
  for (int nt = 0; nt < 8; ++nt)
#pragma unroll
    for (int v = 0; v < 8; ++v) {
      int qr = q0 + v + 8 * hi;
      attn_out[(size_t)(b * 1024 + qr) * U_DIM + h * 128 + nt * 16 + l16] = o[nt][v] * inv[v];
    }
}

// ---------------------------------------------------------------------------
// Kernel 5: residual + LayerNorm (biased var), one block per row.
// ---------------------------------------------------------------------------
__global__ __launch_bounds__(256) void ln_kernel(const float* __restrict__ attn,
                                                 const float* __restrict__ resid,
                                                 const float* __restrict__ gamma,
                                                 const float* __restrict__ beta,
                                                 float* __restrict__ out) {
  const int row = blockIdx.x;
  const int t = threadIdx.x;
  const int lane = t & 31, wid = t >> 5;
  const float* a = attn + (size_t)row * U_DIM;
  const float* r = resid + (size_t)row * U_DIM;
  float x[4], s = 0.f, s2 = 0.f;
#pragma unroll
  for (int i = 0; i < 4; ++i) {
    int c = t + i * 256;
    x[i] = a[c] + r[c];
    s += x[i];
    s2 += x[i] * x[i];
  }
#pragma unroll
  for (int off = 1; off < 32; off <<= 1) { s += __shfl_xor(s, off); s2 += __shfl_xor(s2, off); }
  __shared__ float sh[16];
  if (lane == 0) { sh[wid] = s; sh[8 + wid] = s2; }
  __syncthreads();
  if (t == 0) {
    float ts = 0.f, ts2 = 0.f;
    for (int w = 0; w < 8; ++w) { ts += sh[w]; ts2 += sh[8 + w]; }
    sh[0] = ts;
    sh[8] = ts2;
  }
  __syncthreads();
  float mu = sh[0] * (1.f / 1024.f);
  float var = sh[8] * (1.f / 1024.f) - mu * mu;
  float rr = rsqrtf(var + 1e-5f);
#pragma unroll
  for (int i = 0; i < 4; ++i) {
    int c = t + i * 256;
    out[(size_t)row * U_DIM + c] = (x[i] - mu) * rr * gamma[c] + beta[c];
  }
}

// ---------------------------------------------------------------------------
// Launch
// ---------------------------------------------------------------------------
extern "C" void kernel_launch(void* const* d_in, const int* in_sizes, int n_in,
                              void* d_out, int out_size, void* d_ws, size_t ws_size,
                              hipStream_t stream) {
  const float* queries = (const float*)d_in[0];
  const float* keys    = (const float*)d_in[1];
  const float* values  = (const float*)d_in[2];
  const float* Wq = (const float*)d_in[3];
  const float* bq = (const float*)d_in[4];
  const float* Wk = (const float*)d_in[5];
  const float* bk = (const float*)d_in[6];
  const float* Wv = (const float*)d_in[7];
  const float* bv = (const float*)d_in[8];
  const float* gamma = (const float*)d_in[9];
  const float* beta  = (const float*)d_in[10];

  char* ws = (char*)d_ws;
  const size_t WT_B   = (size_t)U_DIM * U_DIM * 2;   // 2 MB
  const size_t PROJ_B = (size_t)M_DIM * U_DIM * 2;   // 16 MB
  _Float16* Wtq = (_Float16*)(ws);
  _Float16* Wtk = (_Float16*)(ws + WT_B);
  _Float16* Wtv = (_Float16*)(ws + 2 * WT_B);
  _Float16* Qp  = (_Float16*)(ws + 3 * WT_B);
  _Float16* Kp  = (_Float16*)(ws + 3 * WT_B + PROJ_B);
  _Float16* Vt  = (_Float16*)(ws + 3 * WT_B + 2 * PROJ_B);
  float* attn   = (float*)(ws + 3 * WT_B + 3 * PROJ_B);
  float* qmask  = (float*)(ws + 3 * WT_B + 3 * PROJ_B + (size_t)M_DIM * U_DIM * 4);
  float* kmask  = qmask + M_DIM;

  // 1) transpose+convert weights
  wtrans_kernel<<<4096, 256, 0, stream>>>(Wq, Wtq);
  wtrans_kernel<<<4096, 256, 0, stream>>>(Wk, Wtk);
  wtrans_kernel<<<4096, 256, 0, stream>>>(Wv, Wtv);
  // 2) padding masks
  mask_kernel<<<2 * M_DIM, 256, 0, stream>>>(queries, keys, qmask, kmask);
  // 3) projections (V stored transposed), B operand staged via TDM
  dim3 ggrid(U_DIM / 128, M_DIM / 128);
  proj_gemm_kernel<<<ggrid, 256, 0, stream>>>(queries, Wtq, bq, Qp, (long long)U_DIM, 1LL);
  proj_gemm_kernel<<<ggrid, 256, 0, stream>>>(keys,    Wtk, bk, Kp, (long long)U_DIM, 1LL);
  proj_gemm_kernel<<<ggrid, 256, 0, stream>>>(values,  Wtv, bv, Vt, 1LL, (long long)M_DIM);
  // 4) flash attention: 4096 waves = 512 blocks of 8 waves
  flash_attn_kernel<<<512, 256, 0, stream>>>(Qp, Kp, Vt, kmask, qmask, attn);
  // 5) residual + LayerNorm
  ln_kernel<<<M_DIM, 256, 0, stream>>>(attn, queries, gamma, beta, (float*)d_out);
}
